// Attention_13030930776201
// MI455X (gfx1250) — matrix operand order repaired and checked by reading, not compiled, not run on hardware
//
#include <hip/hip_runtime.h>
#include <math.h>

// ---------------- problem constants ----------------
#define BATCH 2
#define TQ    1024
#define SK    1024
#define DMODEL 2048
#define NQH   16
#define NKVH  4
#define HD    128

typedef _Float16 h16;
typedef __attribute__((ext_vector_type(16))) _Float16 v16h;
typedef __attribute__((ext_vector_type(8)))  _Float16 v8h;
typedef __attribute__((ext_vector_type(8)))  float    v8f;
typedef __attribute__((ext_vector_type(4)))  int      i32x4;

// ---------------- WMMA helper ----------------
__device__ __attribute__((always_inline)) inline
v8f wmma_f16(v16h a, v16h b, v8f c) {
  return __builtin_amdgcn_wmma_f32_16x16x32_f16(
      false, a, false, b, (short)0, c, false, false);
}

// A-matrix 16x32 fp16 lane layout: lanes 0-15 hold K=0..7 & 16..23,
// lanes 16-31 hold K=8..15 & 24..31.
__device__ __attribute__((always_inline)) inline
v16h load_a_frag(const h16* p, int lane) {
  int kb = (lane >> 4) << 3;
  v8h lo = *(const v8h*)(p + kb);
  v8h hi = *(const v8h*)(p + kb + 16);
  v16h r;
#pragma unroll
  for (int i = 0; i < 8; ++i) { r[i] = lo[i]; r[i + 8] = hi[i]; }
  return r;
}

// B-matrix 32x16 fp16 lane layout: lanes 0-15 hold K=0..7 & 16..23,
// lanes 16-31 hold K=8..15 & 24..31.
__device__ __attribute__((always_inline)) inline
v16h load_b_frag(const h16* p, int lane) {
  int kb = (lane >> 4) << 3;
  v8h lo = *(const v8h*)(p + kb);
  v8h hi = *(const v8h*)(p + kb + 16);
  v16h r;
#pragma unroll
  for (int i = 0; i < 8; ++i) { r[i] = lo[i]; r[i + 8] = hi[i]; }
  return r;
}

// ---------------- async global->LDS copy (16 bytes per lane) ----------------
__device__ __attribute__((always_inline)) inline
void async_copy16(const h16* gsrc, h16* ldst) {
#if __has_builtin(__builtin_amdgcn_global_load_async_to_lds_b128)
  typedef __attribute__((address_space(1))) i32x4 gv4i;
  typedef __attribute__((address_space(3))) i32x4 lv4i;
  __builtin_amdgcn_global_load_async_to_lds_b128(
      (gv4i*)gsrc, (lv4i*)ldst, 0, 0);
#else
  unsigned loff = (unsigned)(size_t)(__attribute__((address_space(3))) h16*)ldst;
  asm volatile("global_load_async_to_lds_b128 %0, %1, off"
               :: "v"(loff), "v"(gsrc) : "memory");
#endif
}

__device__ __attribute__((always_inline)) inline void wait_async_le8() {
#if __has_builtin(__builtin_amdgcn_s_wait_asynccnt)
  __builtin_amdgcn_s_wait_asynccnt(8);
#else
  asm volatile("s_wait_asynccnt 8" ::: "memory");
#endif
}
__device__ __attribute__((always_inline)) inline void wait_async_le0() {
#if __has_builtin(__builtin_amdgcn_s_wait_asynccnt)
  __builtin_amdgcn_s_wait_asynccnt(0);
#else
  asm volatile("s_wait_asynccnt 0" ::: "memory");
#endif
}

// ---------------- pack kernels ----------------
// fp32 -> fp16, same layout
__global__ void pack_h16_kernel(const float* __restrict__ X, h16* __restrict__ Y, int total)
{
  int idx = blockIdx.x * blockDim.x + threadIdx.x;
  if (idx < total) Y[idx] = (h16)X[idx];
}

// fp32 [K][N] -> fp16 [N][K] tiled transpose (coalesced both sides)
__global__ __launch_bounds__(256) void transpose_pack_kernel(
    const float* __restrict__ W, h16* __restrict__ Wt, int K, int N)
{
  __shared__ float tile[32][33];
  int kb = blockIdx.y * 32, nb = blockIdx.x * 32;
  int tx = threadIdx.x & 31, ty = threadIdx.x >> 5;    // 8 rows of 32
#pragma unroll
  for (int r = ty; r < 32; r += 8)
    tile[r][tx] = W[(size_t)(kb + r) * N + nb + tx];
  __syncthreads();
#pragma unroll
  for (int r = ty; r < 32; r += 8)
    Wt[(size_t)(nb + r) * K + kb + tx] = (h16)tile[tx][r];
}

// ---------------- fp16 GEMM, async double-buffered LDS, fp32 accumulate --------
// C[M,N] = A[M,K] * Bt[N,K]^T   (A row-major, B pre-transposed [N][K])
#define GTM 128
#define GTN 128
#define GTK 64
#define LSTR 72   // padded stride: 144-byte rows keep 16B alignment, spread banks

template <typename CT>
__global__ __launch_bounds__(256) void gemm_h16_async(
    const h16* __restrict__ A, const h16* __restrict__ Bt,
    CT* __restrict__ C, int M, int N, int K)
{
  __shared__ __align__(16) h16 sA[2][GTM * LSTR];
  __shared__ __align__(16) h16 sB[2][GTN * LSTR];

  const int tid  = threadIdx.x;
  const int lane = tid & 31;
  const int wave = tid >> 5;
  const int wm   = wave >> 1;          // 0..3 -> 32 rows
  const int wn   = wave & 1;           // 0..1 -> 64 cols
  const int m0   = blockIdx.y * GTM;
  const int n0   = blockIdx.x * GTN;
  const int lm   = lane & 15;

  v8f acc[2][4];
#pragma unroll
  for (int s = 0; s < 2; ++s)
#pragma unroll
    for (int j = 0; j < 4; ++j) acc[s][j] = (v8f){};

  // stage one 128x64 fp16 tile of A and Bt into buffer b (8 async ops/thread)
  auto issue = [&](int b, int k0) {
#pragma unroll
    for (int i = 0; i < 4; ++i) {
      int idx = tid + 256 * i;
      int row = idx >> 3, c = idx & 7;                  // 8 chunks of 8 h16
      async_copy16(A + (size_t)(m0 + row) * K + k0 + c * 8,
                   &sA[b][row * LSTR + c * 8]);
    }
#pragma unroll
    for (int i = 0; i < 4; ++i) {
      int idx = tid + 256 * i;
      int row = idx >> 3, c = idx & 7;
      async_copy16(Bt + (size_t)(n0 + row) * K + k0 + c * 8,
                   &sB[b][row * LSTR + c * 8]);
    }
  };

  issue(0, 0);
  int cur = 0;
  for (int k0 = 0; k0 < K; k0 += GTK) {
    bool more = (k0 + GTK) < K;
    if (more) issue(cur ^ 1, k0 + GTK);     // prefetch next chunk
    if (more) wait_async_le8();             // older batch (cur) done: in-order
    else      wait_async_le0();
    __syncthreads();

#pragma unroll
    for (int ks = 0; ks < 2; ++ks) {
      v16h af[2], bfr[4];
#pragma unroll
      for (int s = 0; s < 2; ++s)
        af[s] = load_a_frag(&sA[cur][(wm * 32 + s * 16 + lm) * LSTR + ks * 32], lane);
#pragma unroll
      for (int j = 0; j < 4; ++j)
        bfr[j] = load_b_frag(&sB[cur][(wn * 64 + j * 16 + lm) * LSTR + ks * 32], lane);
#pragma unroll
      for (int s = 0; s < 2; ++s)
#pragma unroll
        for (int j = 0; j < 4; ++j)
          acc[s][j] = wmma_f16(af[s], bfr[j], acc[s][j]);
    }
    __syncthreads();                        // all waves done reading buf[cur]
    cur ^= 1;
  }

  const int rbase = (lane >> 4) << 3;
#pragma unroll
  for (int s = 0; s < 2; ++s)
#pragma unroll
    for (int j = 0; j < 4; ++j)
#pragma unroll
      for (int r = 0; r < 8; ++r) {
        int row = m0 + wm * 32 + s * 16 + rbase + r;
        int col = n0 + wn * 64 + j * 16 + lm;
        C[(size_t)row * N + col] = (CT)acc[s][j][r];
      }
}

// ---------------- RoPE + pack fp16 [B*L, NH*H] -> fp16 [B][NH][L][H] ------------
__global__ void rope_pack_kernel(const h16* __restrict__ X,
                                 const int* __restrict__ pos,
                                 h16* __restrict__ Y, int L, int NH)
{
  int idx = blockIdx.x * blockDim.x + threadIdx.x;      // ((b*NH+n)*L+t)*64+hh
  int hh = idx & 63;
  int tmp = idx >> 6;
  int t = tmp % L; tmp /= L;
  int n = tmp % NH; tmp /= NH;
  int b = tmp;
  if (b >= BATCH) return;

  float p = (float)pos[b * L + t];
  float inv_ts = __expf(-(float)hh * (9.210340371976184f / 64.0f)); // 10000^(-hh/64)
  float ang = p * inv_ts;
  float sn, cs;
  __sincosf(ang, &sn, &cs);

  const h16* xr = X + (size_t)(b * L + t) * (NH * HD) + n * HD;
  float x1 = (float)xr[hh], x2 = (float)xr[hh + 64];
  h16* yr = Y + (size_t)((b * NH + n) * L + t) * HD;
  yr[hh]      = (h16)(x1 * cs - x2 * sn);
  yr[hh + 64] = (h16)(x2 * cs + x1 * sn);
}

// ---------------- V pack-transpose: fp16 [B*S, NKV*H] -> fp16 [B][NKV][H][S] ----
__global__ void vpack_kernel(const h16* __restrict__ V, h16* __restrict__ VT)
{
  int idx = blockIdx.x * blockDim.x + threadIdx.x;      // total = 2^20
  int s  = idx & (SK - 1);
  int h  = (idx >> 10) & (HD - 1);
  int kv = (idx >> 17) & (NKVH - 1);
  int b  = idx >> 19;
  if (b >= BATCH) return;
  VT[idx] = V[(size_t)(b * SK + s) * (NKVH * HD) + kv * HD + h];
}

// ---------------- flash attention (fp16 in, fp16 out, fp32 softmax/accum) -------
#define PSTR 72

__global__ __launch_bounds__(128) void attn_kernel(
    const h16* __restrict__ Q, const h16* __restrict__ Km,
    const h16* __restrict__ VT, h16* __restrict__ O)
{
  __shared__ __align__(16) h16 sP[4][16 * PSTR];

  const int lane = threadIdx.x & 31;
  const int wid  = threadIdx.x >> 5;
  const int lm   = lane & 15;
  const int rbase = (lane >> 4) << 3;

  const int ntile = TQ / 64;
  int bid  = blockIdx.x;
  int tile = bid % ntile;
  int nq   = (bid / ntile) % NQH;
  int b    = bid / (ntile * NQH);
  int kv   = nq >> 2;
  int t0   = tile * 64 + wid * 16;

  const h16* qptr = Q  + ((size_t)(b * NQH + nq) * TQ + t0) * HD;
  const h16* kptr = Km + ((size_t)(b * NKVH + kv) * SK) * HD;
  const h16* vptr = VT + ((size_t)(b * NKVH + kv) * HD) * SK;
  h16* myP = &sP[wid][0];

  v16h aq[4];
#pragma unroll
  for (int f = 0; f < 4; ++f)
    aq[f] = load_a_frag(qptr + (size_t)lm * HD + f * 32, lane);

  v8f o[8];
#pragma unroll
  for (int ht = 0; ht < 8; ++ht) o[ht] = (v8f){};
  float mrun[8], lrun[8];
#pragma unroll
  for (int r = 0; r < 8; ++r) { mrun[r] = -INFINITY; lrun[r] = 0.f; }

  for (int s0 = 0; s0 < SK; s0 += 64) {
    float sc[4][8];
#pragma unroll
    for (int jt = 0; jt < 4; ++jt) {
      v8f c = (v8f){};
      const h16* krow = kptr + (size_t)(s0 + jt * 16 + lm) * HD;
#pragma unroll
      for (int f = 0; f < 4; ++f) {
        v16h bk = load_b_frag(krow + f * 32, lane);
        c = wmma_f16(aq[f], bk, c);
      }
#pragma unroll
      for (int r = 0; r < 8; ++r) sc[jt][r] = c[r];
    }

    float mnew[8], alpha[8];
#pragma unroll
    for (int r = 0; r < 8; ++r) {
      float pm = fmaxf(fmaxf(sc[0][r], sc[1][r]), fmaxf(sc[2][r], sc[3][r]));
      pm = fmaxf(pm, __shfl_xor(pm, 1, 32));
      pm = fmaxf(pm, __shfl_xor(pm, 2, 32));
      pm = fmaxf(pm, __shfl_xor(pm, 4, 32));
      pm = fmaxf(pm, __shfl_xor(pm, 8, 32));
      mnew[r]  = fmaxf(mrun[r], pm);
      alpha[r] = __expf(mrun[r] - mnew[r]);
      mrun[r]  = mnew[r];
    }

#pragma unroll
    for (int r = 0; r < 8; ++r) {
      float psum = 0.f;
#pragma unroll
      for (int jt = 0; jt < 4; ++jt) {
        float p = __expf(sc[jt][r] - mnew[r]);
        psum += p;
        myP[(rbase + r) * PSTR + jt * 16 + lm] = (h16)p;
      }
      psum += __shfl_xor(psum, 1, 32);
      psum += __shfl_xor(psum, 2, 32);
      psum += __shfl_xor(psum, 4, 32);
      psum += __shfl_xor(psum, 8, 32);
      lrun[r] = lrun[r] * alpha[r] + psum;
    }
#pragma unroll
    for (int ht = 0; ht < 8; ++ht)
#pragma unroll
      for (int r = 0; r < 8; ++r) o[ht][r] *= alpha[r];

#pragma unroll
    for (int ks = 0; ks < 2; ++ks) {
      v16h ap = load_a_frag(myP + lm * PSTR + ks * 32, lane);
#pragma unroll
      for (int ht = 0; ht < 8; ++ht) {
        v16h bv = load_b_frag(vptr + (size_t)(ht * 16 + lm) * SK + s0 + ks * 32, lane);
        o[ht] = wmma_f16(ap, bv, o[ht]);
      }
    }
  }

  float inv[8];
#pragma unroll
  for (int r = 0; r < 8; ++r) inv[r] = 1.f / lrun[r];
#pragma unroll
  for (int ht = 0; ht < 8; ++ht)
#pragma unroll
    for (int r = 0; r < 8; ++r) {
      int row = b * TQ + t0 + rbase + r;
      int col = nq * HD + ht * 16 + lm;
      O[(size_t)row * (NQH * HD) + col] = (h16)(o[ht][r] * inv[r]);
    }
}

// ---------------- workspace layout (bytes) ----------------
#define SZ_2048x2048_H ((size_t)2048 * 2048 * 2)   // 8,388,608
#define SZ_2048x512_H  ((size_t)2048 * 512 * 2)    // 2,097,152
#define OFF_XQH   ((size_t)0)
#define OFF_XKVH  (OFF_XQH  + SZ_2048x2048_H)
#define OFF_WQT   (OFF_XKVH + SZ_2048x2048_H)
#define OFF_WKT   (OFF_WQT  + SZ_2048x2048_H)
#define OFF_WVT   (OFF_WKT  + SZ_2048x512_H)
#define OFF_WOT   (OFF_WVT  + SZ_2048x512_H)
#define OFF_QF    (OFF_WOT  + SZ_2048x2048_H)
#define OFF_KF    (OFF_QF   + SZ_2048x2048_H)
#define OFF_VF    (OFF_KF   + SZ_2048x512_H)
#define OFF_QH    (OFF_VF   + SZ_2048x512_H)
#define OFF_KH    (OFF_QH   + SZ_2048x2048_H)
#define OFF_VTH   (OFF_KH   + SZ_2048x512_H)
#define OFF_ATTH  (OFF_VTH  + SZ_2048x512_H)
// total ~71.3 MB

extern "C" void kernel_launch(void* const* d_in, const int* in_sizes, int n_in,
                              void* d_out, int out_size, void* d_ws, size_t ws_size,
                              hipStream_t stream) {
  const float* Xq   = (const float*)d_in[0];
  const float* Xkv  = (const float*)d_in[1];
  const int*   qpos = (const int*)d_in[2];
  const int*   kpos = (const int*)d_in[3];
  const float* Wq   = (const float*)d_in[4];
  const float* Wk   = (const float*)d_in[5];
  const float* Wv   = (const float*)d_in[6];
  const float* Wo   = (const float*)d_in[7];
  float* out = (float*)d_out;

  char* ws = (char*)d_ws;
  h16* xqh  = (h16*)(ws + OFF_XQH);
  h16* xkvh = (h16*)(ws + OFF_XKVH);
  h16* wqt  = (h16*)(ws + OFF_WQT);
  h16* wkt  = (h16*)(ws + OFF_WKT);
  h16* wvt  = (h16*)(ws + OFF_WVT);
  h16* wot  = (h16*)(ws + OFF_WOT);
  h16* qf   = (h16*)(ws + OFF_QF);
  h16* kf   = (h16*)(ws + OFF_KF);
  h16* vf   = (h16*)(ws + OFF_VF);
  h16* qh   = (h16*)(ws + OFF_QH);
  h16* kh   = (h16*)(ws + OFF_KH);
  h16* vth  = (h16*)(ws + OFF_VTH);
  h16* atth = (h16*)(ws + OFF_ATTH);

  const int M  = BATCH * TQ;        // 2048
  const int Nq = NQH * HD;          // 2048
  const int Nk = NKVH * HD;         // 512

  // one-time fp16 packing: activations same-layout, weights transposed [N][K]
  pack_h16_kernel<<<(M * DMODEL) / 256, 256, 0, stream>>>(Xq,  xqh,  M * DMODEL);
  pack_h16_kernel<<<(M * DMODEL) / 256, 256, 0, stream>>>(Xkv, xkvh, M * DMODEL);
  transpose_pack_kernel<<<dim3(Nq / 32, DMODEL / 32), 256, 0, stream>>>(Wq, wqt, DMODEL, Nq);
  transpose_pack_kernel<<<dim3(Nk / 32, DMODEL / 32), 256, 0, stream>>>(Wk, wkt, DMODEL, Nk);
  transpose_pack_kernel<<<dim3(Nk / 32, DMODEL / 32), 256, 0, stream>>>(Wv, wvt, DMODEL, Nk);
  transpose_pack_kernel<<<dim3(DMODEL / 32, Nq / 32), 256, 0, stream>>>(Wo, wot, Nq, DMODEL);

  // QKV projections (fp16 WMMA, async double-buffered staging)
  gemm_h16_async<h16><<<dim3(Nq / GTN, M / GTM), 256, 0, stream>>>(xqh,  wqt, qf, M, Nq, DMODEL);
  gemm_h16_async<h16><<<dim3(Nk / GTN, M / GTM), 256, 0, stream>>>(xkvh, wkt, kf, M, Nk, DMODEL);
  gemm_h16_async<h16><<<dim3(Nk / GTN, M / GTM), 256, 0, stream>>>(xkvh, wvt, vf, M, Nk, DMODEL);

  // RoPE + head-major pack
  rope_pack_kernel<<<(BATCH * NQH  * TQ * 64) / 256, 256, 0, stream>>>(qf, qpos, qh, TQ, NQH);
  rope_pack_kernel<<<(BATCH * NKVH * SK * 64) / 256, 256, 0, stream>>>(kf, kpos, kh, SK, NKVH);
  vpack_kernel<<<(BATCH * NKVH * HD * SK) / 256, 256, 0, stream>>>(vf, vth);

  // flash attention
  attn_kernel<<<BATCH * NQH * (TQ / 64), 128, 0, stream>>>(qh, kh, vth, atth);

  // output projection -> fp32 d_out
  gemm_h16_async<float><<<dim3(DMODEL / GTN, M / GTM), 256, 0, stream>>>(atth, wot, out, M, DMODEL, Nq);
}
